// SinkhornLoss_7808250544443
// MI455X (gfx1250) — matrix operand order repaired and checked by reading, not compiled, not run on hardware
//
#include <hip/hip_runtime.h>
#include <hip/hip_bf16.h>
#include <hip/hip_fp16.h>

#define B_N 4096
#define D_K 512
#define EPS 0.1f
#define INV_EPS 10.0f
#define N_ITER 50

typedef __attribute__((ext_vector_type(16))) __bf16    v16bf;
typedef __attribute__((ext_vector_type(8)))  __bf16    v8bf;
typedef __attribute__((ext_vector_type(8)))  float     v8f;
typedef __attribute__((ext_vector_type(4)))  _Float16  v4h;
typedef __attribute__((ext_vector_type(4)))  float     v4f;

// ---------------------------------------------------------------------------
// f32 -> bf16 conversion of one 512-wide row + squared-norm of the row.
__global__ void prep_kernel(const float* __restrict__ src,
                            __bf16* __restrict__ dst,
                            float* __restrict__ norms) {
    __shared__ float red[128];
    const int row = blockIdx.x;
    const int tid = threadIdx.x;
    const float* r = src + row * D_K;
    __bf16* w = dst + row * D_K;
    float acc = 0.0f;
    #pragma unroll
    for (int j = tid; j < D_K; j += 128) {
        float v = r[j];
        acc += v * v;
        w[j] = (__bf16)v;
    }
    red[tid] = acc;
    __syncthreads();
    for (int off = 64; off > 0; off >>= 1) {
        if (tid < off) red[tid] += red[tid + off];
        __syncthreads();
    }
    if (tid == 0) norms[row] = red[0];
}

__global__ void zero_kernel(float* __restrict__ p, int n) {
    int i = blockIdx.x * blockDim.x + threadIdx.x;
    if (i < n) p[i] = 0.0f;
}

__device__ __forceinline__ v16bf load_frag(const __bf16* p0, const __bf16* p1) {
    v8bf lo = *(const v8bf*)p0;
    v8bf hi = *(const v8bf*)p1;
    v16bf r;
    #pragma unroll
    for (int e = 0; e < 8; ++e) { r[e] = lo[e]; r[8 + e] = hi[e]; }
    return r;
}

// ---------------------------------------------------------------------------
// Cost matrix via bf16 WMMA, stored f16:
//   Ch[i][j] = (f16) max(x2[i] + y2[j] - 2 * (X Y^T)[i][j], 0)
// Each wave computes a 16(M) x 64(N) strip; fragments are register
// double-buffered so next-chunk loads overlap the current WMMAs.
__global__ void gemm_cost_kernel(const __bf16* __restrict__ Xb,
                                 const __bf16* __restrict__ Yb,
                                 const float* __restrict__ x2,
                                 const float* __restrict__ y2,
                                 _Float16* __restrict__ Ch) {
    const int wave = (blockIdx.x * blockDim.x + threadIdx.x) >> 5;
    const int lane = threadIdx.x & 31;
    const int m0   = (wave >> 6) << 4;   // 256 row tiles of 16
    const int n0   = (wave & 63) << 6;   // 64 col groups of 64
    const int half = lane >> 4;
    const int l16  = lane & 15;

    v8f acc[4] = {v8f{}, v8f{}, v8f{}, v8f{}};

    const __bf16* arow = Xb + (m0 + l16) * D_K;
    const __bf16* b0   = Yb + (n0 + l16) * D_K + 16 * half;
    const __bf16* b1   = b0 + 16 * D_K;
    const __bf16* b2   = b0 + 32 * D_K;
    const __bf16* b3   = b0 + 48 * D_K;

    v16bf a_cur = load_frag(arow + 16 * half, arow + 16 * half + 8);
    v16bf bc0 = load_frag(b0, b0 + 8);
    v16bf bc1 = load_frag(b1, b1 + 8);
    v16bf bc2 = load_frag(b2, b2 + 8);
    v16bf bc3 = load_frag(b3, b3 + 8);

    #pragma unroll
    for (int kc = 0; kc < D_K; kc += 32) {
        v16bf a_nxt, bn0, bn1, bn2, bn3;
        if (kc + 32 < D_K) {
            const int kn = kc + 32;
            a_nxt = load_frag(arow + kn + 16 * half, arow + kn + 16 * half + 8);
            bn0 = load_frag(b0 + kn, b0 + kn + 8);
            bn1 = load_frag(b1 + kn, b1 + kn + 8);
            bn2 = load_frag(b2 + kn, b2 + kn + 8);
            bn3 = load_frag(b3 + kn, b3 + kn + 8);
        }
        acc[0] = __builtin_amdgcn_wmma_f32_16x16x32_bf16(false, a_cur, false, bc0, (short)0, acc[0], false, false);
        acc[1] = __builtin_amdgcn_wmma_f32_16x16x32_bf16(false, a_cur, false, bc1, (short)0, acc[1], false, false);
        acc[2] = __builtin_amdgcn_wmma_f32_16x16x32_bf16(false, a_cur, false, bc2, (short)0, acc[2], false, false);
        acc[3] = __builtin_amdgcn_wmma_f32_16x16x32_bf16(false, a_cur, false, bc3, (short)0, acc[3], false, false);
        a_cur = a_nxt; bc0 = bn0; bc1 = bn1; bc2 = bn2; bc3 = bn3;
    }

    float xr[8];
    #pragma unroll
    for (int v = 0; v < 8; ++v) xr[v] = x2[m0 + half * 8 + v];

    #pragma unroll
    for (int t = 0; t < 4; ++t) {
        const int col = n0 + t * 16 + l16;
        const float yv = y2[col];
        #pragma unroll
        for (int v = 0; v < 8; ++v) {
            const int row = m0 + half * 8 + v;
            float c = xr[v] + yv - 2.0f * acc[t][v];
            Ch[(size_t)row * B_N + col] = (_Float16)fmaxf(c, 0.0f);
        }
    }
}

// ---------------------------------------------------------------------------
// 64x64 LDS-tiled f16 transpose: dst[c][r] = src[r][c].
// grid = 4096 blocks (64x64 grid of tiles), 256 threads.
__global__ void transpose_kernel(const _Float16* __restrict__ src,
                                 _Float16* __restrict__ dst) {
    __shared__ _Float16 tile[64][68];   // pad 8B to spread banks
    const int bx = blockIdx.x & 63;     // tile col
    const int by = blockIdx.x >> 6;     // tile row
    const int tx = threadIdx.x & 15;    // 16 * 4 halves = 64 cols
    const int ty = threadIdx.x >> 4;    // 16 rows per pass
    #pragma unroll
    for (int p = 0; p < 4; ++p) {
        const int r = ty + p * 16;
        v4h d = *(const v4h*)(src + (size_t)(by * 64 + r) * B_N + bx * 64 + tx * 4);
        #pragma unroll
        for (int k = 0; k < 4; ++k) tile[r][tx * 4 + k] = d[k];
    }
    __syncthreads();
    #pragma unroll
    for (int p = 0; p < 4; ++p) {
        const int rr = ty + p * 16;     // dst row within tile = src col
        v4h d;
        #pragma unroll
        for (int k = 0; k < 4; ++k) d[k] = tile[tx * 4 + k][rr];
        *(v4h*)(dst + (size_t)(bx * 64 + rr) * B_N + by * 64 + tx * 4) = d;
    }
}

// ---------------------------------------------------------------------------
// out_vec[i] = -eps * LSE_j( -M[i][j]/eps + in_vec[j] )  with +1e-10 inside log.
// Generic over row-major f16 matrix; used for both axes (col pass uses ChT).
// grid = 4096 blocks x 256 threads; v4h cost loads + float4 potential loads.
__global__ void lse_kernel(const _Float16* __restrict__ M,
                           const float* __restrict__ in_vec,
                           float* __restrict__ out_vec) {
    __shared__ float sm[256];
    __shared__ float ss[256];
    const int i = blockIdx.x, tid = threadIdx.x;
    const _Float16* row = M + (size_t)i * B_N;
    float m = -3.4e38f, s = 0.0f;
    for (int j = tid * 4; j < B_N; j += 1024) {
        v4h c4  = *(const v4h*)(row + j);
        v4f lb4 = *(const v4f*)(in_vec + j);
        #pragma unroll
        for (int k = 0; k < 4; ++k) {
            float v = (float)c4[k] * (-INV_EPS) + lb4[k];
            float Mx = fmaxf(m, v);
            s = s * __expf(m - Mx) + __expf(v - Mx);
            m = Mx;
        }
    }
    sm[tid] = m; ss[tid] = s;
    __syncthreads();
    for (int off = 128; off > 0; off >>= 1) {
        if (tid < off) {
            float m2 = sm[tid + off], s2 = ss[tid + off];
            float Mx = fmaxf(sm[tid], m2);
            ss[tid] = ss[tid] * __expf(sm[tid] - Mx) + s2 * __expf(m2 - Mx);
            sm[tid] = Mx;
        }
        __syncthreads();
    }
    if (tid == 0) out_vec[i] = -EPS * (sm[0] + __logf(ss[0] + 1e-10f));
}

// ---------------------------------------------------------------------------
// partial[i] = sum_j exp(-C/eps + la[i] + lb[j]) * C[i][j]   (C from f16 mirror)
__global__ void loss_row_kernel(const _Float16* __restrict__ Ch,
                                const float* __restrict__ la,
                                const float* __restrict__ lb,
                                float* __restrict__ partial) {
    __shared__ float red[256];
    const int i = blockIdx.x, tid = threadIdx.x;
    const _Float16* row = Ch + (size_t)i * B_N;
    const float lai = la[i];
    float acc = 0.0f;
    for (int j = tid * 4; j < B_N; j += 1024) {
        v4h c4  = *(const v4h*)(row + j);
        v4f lb4 = *(const v4f*)(lb + j);
        #pragma unroll
        for (int k = 0; k < 4; ++k) {
            float c = (float)c4[k];
            acc += __expf(c * (-INV_EPS) + lai + lb4[k]) * c;
        }
    }
    red[tid] = acc;
    __syncthreads();
    for (int off = 128; off > 0; off >>= 1) {
        if (tid < off) red[tid] += red[tid + off];
        __syncthreads();
    }
    if (tid == 0) partial[i] = red[0];
}

__global__ void final_reduce_kernel(const float* __restrict__ partial,
                                    float* __restrict__ out) {
    __shared__ float red[1024];
    const int tid = threadIdx.x;
    float acc = 0.0f;
    for (int j = tid; j < B_N; j += 1024) acc += partial[j];
    red[tid] = acc;
    __syncthreads();
    for (int off = 512; off > 0; off >>= 1) {
        if (tid < off) red[tid] += red[tid + off];
        __syncthreads();
    }
    if (tid == 0) out[0] = red[0] * (1.0f / (float)B_N);
}

// ---------------------------------------------------------------------------
extern "C" void kernel_launch(void* const* d_in, const int* in_sizes, int n_in,
                              void* d_out, int out_size, void* d_ws, size_t ws_size,
                              hipStream_t stream) {
    const float* real = (const float*)d_in[0];
    const float* fake = (const float*)d_in[1];

    char* ws = (char*)d_ws;
    _Float16* Ch  = (_Float16*)(ws);                    // 32 MB
    _Float16* ChT = (_Float16*)(ws + 33554432);         // 32 MB
    __bf16*   Xb  = (__bf16*)(ws + 67108864);           // 4 MB
    __bf16*   Yb  = (__bf16*)(ws + 71303168);           // 4 MB
    float*    x2      = (float*)(ws + 75497472);
    float*    y2      = (float*)(ws + 75497472 + 1 * 16384);
    float*    la      = (float*)(ws + 75497472 + 2 * 16384);
    float*    lb      = (float*)(ws + 75497472 + 3 * 16384);
    float*    partial = (float*)(ws + 75497472 + 4 * 16384);

    prep_kernel<<<B_N, 128, 0, stream>>>(real, Xb, x2);
    prep_kernel<<<B_N, 128, 0, stream>>>(fake, Yb, y2);
    zero_kernel<<<16, 256, 0, stream>>>(lb, B_N);

    gemm_cost_kernel<<<2048, 256, 0, stream>>>(Xb, Yb, x2, y2, Ch);
    transpose_kernel<<<4096, 256, 0, stream>>>(Ch, ChT);

    for (int it = 0; it < N_ITER; ++it) {
        lse_kernel<<<B_N, 256, 0, stream>>>(Ch,  lb, la);   // rows  -> log_a
        lse_kernel<<<B_N, 256, 0, stream>>>(ChT, la, lb);   // cols  -> log_b
    }

    loss_row_kernel<<<B_N, 256, 0, stream>>>(Ch, la, lb, partial);
    final_reduce_kernel<<<1, 1024, 0, stream>>>(partial, (float*)d_out);
}
